// MultiHeadSelfAttentionBlock_35235911696676
// MI455X (gfx1250) — matrix operand order repaired and checked by reading, not compiled, not run on hardware
//
#include <hip/hip_runtime.h>
#include <stdint.h>

// ---------------------------------------------------------------------------
// MHSA block (EfficientViT-style) for MI455X / gfx1250, bf16 WMMA pipeline.
// ---------------------------------------------------------------------------

#define B_  32
#define C_  640
#define H_  32
#define W_  32
#define NH_ 8
#define KD_ 64
#define VD_ 64
#define L_  1024   // H*W
#define P_  256    // 16*16 after stride-2 dwconv

typedef __attribute__((ext_vector_type(16))) __bf16 v16bf;
typedef __attribute__((ext_vector_type(8)))  float  v8f;

union Frag {
    v16bf v;
    uint4 q[2];
    unsigned short u[16];
};
union H8 {
    uint4 q;
    _Float16 h[8];
};

__device__ __forceinline__ unsigned short f2bf(float f) {
    union { float f; unsigned u; } x; x.f = f;
    unsigned u = x.u;
    unsigned r = (u + 0x7FFFu + ((u >> 16) & 1u)) >> 16;  // RNE
    return (unsigned short)r;
}
__device__ __forceinline__ float bf2f(unsigned short h) {
    union { unsigned u; float f; } x; x.u = ((unsigned)h) << 16; return x.f;
}
__device__ __forceinline__ v8f zero8() {
    v8f z = {0.f, 0.f, 0.f, 0.f, 0.f, 0.f, 0.f, 0.f};
    return z;
}
__device__ __forceinline__ v8f wmma_bf16(const Frag &a, const Frag &b, v8f c) {
    return __builtin_amdgcn_wmma_f32_16x16x32_bf16(
        /*neg_a=*/false, a.v, /*neg_b=*/false, b.v,
        /*c_mod=*/(short)0, c, /*reuse_a=*/false, /*reuse_b=*/false);
}

// ---------------------------------------------------------------------------
// Kernel 1: input BatchNorm -> xn (bf16)
// ---------------------------------------------------------------------------
__global__ void bn_in_kernel(const float* __restrict__ x,
                             const float* __restrict__ gamma,
                             const float* __restrict__ beta,
                             const float* __restrict__ mean,
                             const float* __restrict__ var,
                             unsigned short* __restrict__ xn) {
    long long i = (long long)blockIdx.x * blockDim.x + threadIdx.x;
    const long long N = (long long)B_ * C_ * H_ * W_;
    if (i >= N) return;
    int c = (int)((i / (H_ * W_)) % C_);
    float s = gamma[c] * rsqrtf(var[c] + 1e-3f);
    xn[i] = f2bf((x[i] - mean[c]) * s + beta[c]);
}

// ---------------------------------------------------------------------------
// Kernel 2: fused stride-2 depthwise 3x3 (k & v paths share input taps) + BN
// outputs kf, vf: [B][C][16][16] bf16
// ---------------------------------------------------------------------------
__global__ void dwconv_kernel(const unsigned short* __restrict__ xn,
                              const float* __restrict__ kw, const float* __restrict__ kg,
                              const float* __restrict__ kb, const float* __restrict__ km,
                              const float* __restrict__ kv,
                              const float* __restrict__ vw, const float* __restrict__ vg,
                              const float* __restrict__ vb, const float* __restrict__ vm,
                              const float* __restrict__ vv,
                              unsigned short* __restrict__ kf,
                              unsigned short* __restrict__ vf) {
    int i = blockIdx.x * blockDim.x + threadIdx.x;   // over B*C*256
    if (i >= B_ * C_ * 256) return;
    int ox = i & 15, oy = (i >> 4) & 15;
    int c = (i >> 8) % C_;
    int b = i / (C_ * 256);
    const unsigned short* xp = xn + (long long)(b * C_ + c) * (H_ * W_);
    float acck = 0.f, accv = 0.f;
    #pragma unroll
    for (int ky = 0; ky < 3; ++ky) {
        int iy = oy * 2 - 1 + ky;
        if (iy < 0 || iy >= H_) continue;
        #pragma unroll
        for (int kx = 0; kx < 3; ++kx) {
            int ix = ox * 2 - 1 + kx;
            if (ix < 0 || ix >= W_) continue;
            float xv = bf2f(xp[iy * W_ + ix]);
            acck += xv * kw[c * 9 + ky * 3 + kx];
            accv += xv * vw[c * 9 + ky * 3 + kx];
        }
    }
    float sk = kg[c] * rsqrtf(kv[c] + 1e-3f);
    float sv = vg[c] * rsqrtf(vv[c] + 1e-3f);
    kf[i] = f2bf((acck - km[c]) * sk + kb[c]);
    vf[i] = f2bf((accv - vm[c]) * sv + vb[c]);
}

// ---------------------------------------------------------------------------
// Kernel 3: generic batched WMMA GEMM  C[b] = A(fp32 MxK) * B[b](bf16 KxN)
// 128x128 workgroup tile, 8 waves, each wave 32x64 (2x4 WMMA subtiles).
// MODE 0: bf16 C row-major [b][M][N]
// MODE 1: bf16 C transposed [b][N][M]
// MODE 2: fp32 out = acc*ls[col&31] + shortcut, row-major [b][M][N]
// ---------------------------------------------------------------------------
template <int MODE>
__global__ __launch_bounds__(256)
void gemm_kernel(const float* __restrict__ A,
                 const unsigned short* __restrict__ Bm,
                 unsigned short* __restrict__ Cb,
                 float* __restrict__ Cf,
                 const float* __restrict__ ls,
                 const float* __restrict__ shortcut,
                 int M, int N, int K) {
    __shared__ unsigned short As[128 * 32];   // [m][k]
    __shared__ unsigned short Bs[128 * 32];   // [n][k]  (transposed in LDS)
    int tid = threadIdx.x;
    int b = blockIdx.z;
    int bm = blockIdx.y * 128;
    int bn = blockIdx.x * 128;
    const unsigned short* Bb = Bm + (long long)b * K * N;
    int wv = tid >> 5, lane = tid & 31;
    int wm = wv & 3, wn = wv >> 2;
    int lm = lane & 15, half = lane >> 4;

    v8f acc[2][4];
    #pragma unroll
    for (int i = 0; i < 2; ++i)
        #pragma unroll
        for (int j = 0; j < 4; ++j) acc[i][j] = zero8();

    for (int k0 = 0; k0 < K; k0 += 32) {
        // Stage A tile (fp32 -> bf16)
        #pragma unroll
        for (int it = 0; it < 16; ++it) {
            int lin = it * 256 + tid;
            int r = lin >> 5, cc = lin & 31;
            float v = 0.f;
            if (bm + r < M) v = A[(long long)(bm + r) * K + k0 + cc];
            As[r * 32 + cc] = f2bf(v);
        }
        // Stage B tile transposed: Bs[n][k]
        #pragma unroll
        for (int it = 0; it < 16; ++it) {
            int lin = it * 256 + tid;
            int kk = lin >> 7, nn = lin & 127;
            Bs[nn * 32 + kk] = Bb[(long long)(k0 + kk) * N + bn + nn];
        }
        __syncthreads();

        Frag a[2];
        #pragma unroll
        for (int mi = 0; mi < 2; ++mi) {
            const unsigned short* p = &As[(wm * 32 + mi * 16 + lm) * 32 + half * 8];
            a[mi].q[0] = *(const uint4*)(p);
            a[mi].q[1] = *(const uint4*)(p + 16);
        }
        #pragma unroll
        for (int ni = 0; ni < 4; ++ni) {
            Frag bf;
            const unsigned short* p = &Bs[(wn * 64 + ni * 16 + lm) * 32 + half * 8];
            bf.q[0] = *(const uint4*)(p);
            bf.q[1] = *(const uint4*)(p + 16);
            #pragma unroll
            for (int mi = 0; mi < 2; ++mi)
                acc[mi][ni] = wmma_bf16(a[mi], bf, acc[mi][ni]);
        }
        __syncthreads();
    }

    // Epilogue
    #pragma unroll
    for (int mi = 0; mi < 2; ++mi) {
        #pragma unroll
        for (int ni = 0; ni < 4; ++ni) {
            int col = bn + wn * 64 + ni * 16 + lm;
            #pragma unroll
            for (int r = 0; r < 8; ++r) {
                int row = bm + wm * 32 + mi * 16 + half * 8 + r;
                if (row < M) {
                    float v = acc[mi][ni][r];
                    if (MODE == 0) {
                        Cb[(long long)b * M * N + (long long)row * N + col] = f2bf(v);
                    } else if (MODE == 1) {
                        Cb[(long long)b * M * N + (long long)col * M + row] = f2bf(v);
                    } else {
                        long long idx = (long long)b * M * N + (long long)row * N + col;
                        Cf[idx] = v * ls[col & 31] + shortcut[idx];
                    }
                }
            }
        }
    }
}

// ---------------------------------------------------------------------------
// Kernel 4: attention, two-pass softmax to keep register pressure low
// (round 1 version spilled 16 v8f accumulators to scratch).
// Block = 128 threads (4 waves); each wave owns 16 L-rows of one (b, nh).
// Pass 1: S = Q K^T one 16x16 tile at a time (single live accumulator),
//         raw logits -> wave-private LDS (f16), track per-row max.
// Pass 2: rebuild probs on the fly from LDS (exp in fp32, accumulate row
//         sums, pack bf16 A-fragments in registers), O = P V via WMMA,
//         divide by row sums in the epilogue.
//   qproj : [B][512][1024] bf16   (torch-view scramble resolved at load)
//   kmat  : [B][P=256][KD=64] bf16 (B-operand friendly: lane=col p, K contig)
//   vmat  : [B][VD=64][P=256] bf16 (lane=col vd, K=p contig)
//   omat  : [B][512][1024] bf16, channel c' = nh*64+vd, spatial l
// ---------------------------------------------------------------------------
__global__ __launch_bounds__(128)
void attn_kernel(const unsigned short* __restrict__ qproj,
                 const unsigned short* __restrict__ kmat,
                 const unsigned short* __restrict__ vmat,
                 unsigned short* __restrict__ omat) {
    __shared__ _Float16 slds[4][16 * 256];   // per-wave raw logits [16 rows][256 cols]
    __shared__ float rowmax[4][16];
    __shared__ float rowsum[4][16];

    int tid = threadIdx.x;
    int wv = tid >> 5, lane = tid & 31;
    int lm = lane & 15, half = lane >> 4;
    int b = blockIdx.z, nh = blockIdx.y;
    int lbase = blockIdx.x * 64 + wv * 16;

    // Q A-fragments: lane's row m = lm, l = lbase+lm.
    // Faithful torch .view scramble: o = nh*64 + 2h + (w>=16), offset 64*(w%16).
    int l = lbase + lm;
    int h = l >> 5, w = l & 31;
    int o = nh * 64 + 2 * h + (w >> 4);
    const unsigned short* qrow =
        qproj + (long long)(b * 512 + o) * 1024 + 64 * (w & 15);
    Frag qa[2];
    #pragma unroll
    for (int ck = 0; ck < 2; ++ck) {
        int K0 = ck * 32 + half * 8;
        qa[ck].q[0] = *(const uint4*)(qrow + K0);
        qa[ck].q[1] = *(const uint4*)(qrow + K0 + 16);
    }

    const unsigned short* kb = kmat + (long long)b * P_ * KD_;
    const unsigned short* vb = vmat + (long long)b * VD_ * P_;
    _Float16* sw = slds[wv];

    // ---- Pass 1: S tiles -> LDS raw logits, track per-row max -------------
    float mx[8];
    #pragma unroll
    for (int r = 0; r < 8; ++r) mx[r] = -1e30f;

    #pragma unroll 4
    for (int t = 0; t < 16; ++t) {
        v8f sacc = zero8();
        int p = t * 16 + lm;  // this lane's column (key index)
        #pragma unroll
        for (int ck = 0; ck < 2; ++ck) {
            Frag kfr;
            const unsigned short* kp = kb + p * KD_ + ck * 32 + half * 8;
            kfr.q[0] = *(const uint4*)(kp);
            kfr.q[1] = *(const uint4*)(kp + 16);
            sacc = wmma_bf16(qa[ck], kfr, sacc);
        }
        #pragma unroll
        for (int r = 0; r < 8; ++r) {
            float v = sacc[r];
            mx[r] = fmaxf(mx[r], v);
            sw[(half * 8 + r) * 256 + t * 16 + lm] = (_Float16)v;
        }
    }
    // Row max across the 16 lanes holding each row, publish by row index.
    #pragma unroll
    for (int r = 0; r < 8; ++r) {
        float m = mx[r];
        #pragma unroll
        for (int d = 1; d < 16; d <<= 1) m = fmaxf(m, __shfl_xor(m, d, 32));
        if (lane == r)      rowmax[wv][r]     = m;   // rows 0..7  (half 0)
        if (lane == 16 + r) rowmax[wv][8 + r] = m;   // rows 8..15 (half 1)
    }
    __syncthreads();

    // ---- Pass 2: O = softmax(S) V, probs built on the fly -----------------
    float rmx = rowmax[wv][lm];   // max of this lane's A-fragment row
    v8f oa[4];
    #pragma unroll
    for (int ni = 0; ni < 4; ++ni) oa[ni] = zero8();
    float psum = 0.f;

    #pragma unroll 2
    for (int ck = 0; ck < 8; ++ck) {
        // A fragment for rows=lm: K positions {half*8..+7, half*8+16..+23} of chunk
        const _Float16* pp = &sw[lm * 256 + ck * 32 + half * 8];
        H8 h0, h1;
        h0.q = *(const uint4*)(pp);
        h1.q = *(const uint4*)(pp + 16);
        Frag pa;
        #pragma unroll
        for (int i = 0; i < 8; ++i) {
            float p0 = __expf(((float)h0.h[i] - rmx) * 0.125f);
            float p1 = __expf(((float)h1.h[i] - rmx) * 0.125f);
            psum += p0 + p1;
            pa.u[i]     = f2bf(p0);
            pa.u[8 + i] = f2bf(p1);
        }
        #pragma unroll
        for (int ni = 0; ni < 4; ++ni) {
            Frag vfg;
            const unsigned short* vp = vb + (ni * 16 + lm) * P_ + ck * 32 + half * 8;
            vfg.q[0] = *(const uint4*)(vp);
            vfg.q[1] = *(const uint4*)(vp + 16);
            oa[ni] = wmma_bf16(pa, vfg, oa[ni]);
        }
    }
    // Combine the two half-wave K partitions of each row's sum, publish.
    psum += __shfl_xor(psum, 16, 32);
    if (half == 0) rowsum[wv][lm] = psum;
    __syncthreads();

    // ---- Normalize and scatter to omat[b][nh*64+vd][l] --------------------
    #pragma unroll
    for (int ni = 0; ni < 4; ++ni) {
        int cd = nh * 64 + ni * 16 + lm;
        #pragma unroll
        for (int r = 0; r < 8; ++r) {
            int lrow = lbase + half * 8 + r;
            float v = oa[ni][r] / rowsum[wv][half * 8 + r];
            omat[(long long)(b * 512 + cd) * 1024 + lrow] = f2bf(v);
        }
    }
}

// ---------------------------------------------------------------------------
// Host-side launcher
// ---------------------------------------------------------------------------
extern "C" void kernel_launch(void* const* d_in, const int* in_sizes, int n_in,
                              void* d_out, int out_size, void* d_ws, size_t ws_size,
                              hipStream_t stream) {
    const float* x        = (const float*)d_in[0];
    const float* in_g     = (const float*)d_in[1];
    const float* in_b     = (const float*)d_in[2];
    const float* in_m     = (const float*)d_in[3];
    const float* in_v     = (const float*)d_in[4];
    const float* q_w      = (const float*)d_in[5];
    const float* k_dw_w   = (const float*)d_in[6];
    const float* k_g      = (const float*)d_in[7];
    const float* k_b      = (const float*)d_in[8];
    const float* k_m      = (const float*)d_in[9];
    const float* k_v      = (const float*)d_in[10];
    const float* k_w      = (const float*)d_in[11];
    const float* v_dw_w   = (const float*)d_in[12];
    const float* v_g      = (const float*)d_in[13];
    const float* v_b      = (const float*)d_in[14];
    const float* v_m      = (const float*)d_in[15];
    const float* v_v      = (const float*)d_in[16];
    const float* v_w      = (const float*)d_in[17];
    const float* out_w    = (const float*)d_in[18];
    const float* ls       = (const float*)d_in[19];
    float* out            = (float*)d_out;

    // Workspace partition (bf16 elements, ~132 MB total)
    unsigned short* xn    = (unsigned short*)d_ws;
    unsigned short* kf    = xn    + (size_t)B_ * C_ * H_ * W_;
    unsigned short* vf    = kf    + (size_t)B_ * C_ * P_;
    unsigned short* kmat  = vf    + (size_t)B_ * C_ * P_;
    unsigned short* vmat  = kmat  + (size_t)B_ * P_ * KD_;
    unsigned short* qproj = vmat  + (size_t)B_ * VD_ * P_;
    unsigned short* omat  = qproj + (size_t)B_ * 512 * L_;

    // 1) input BN -> bf16
    {
        long long n = (long long)B_ * C_ * H_ * W_;
        bn_in_kernel<<<(unsigned)((n + 255) / 256), 256, 0, stream>>>(
            x, in_g, in_b, in_m, in_v, xn);
    }
    // 2) fused dwconv(stride 2) + BN for k & v
    {
        int n = B_ * C_ * P_;
        dwconv_kernel<<<(n + 255) / 256, 256, 0, stream>>>(
            xn, k_dw_w, k_g, k_b, k_m, k_v,
            v_dw_w, v_g, v_b, v_m, v_v, kf, vf);
    }
    // 3) k projection: [64x640] x [640x256] -> stored transposed [P][KD]
    gemm_kernel<1><<<dim3(2, 1, B_), 256, 0, stream>>>(
        k_w, kf, kmat, nullptr, nullptr, nullptr, KD_, P_, C_);
    // 4) v projection: stored natural [VD][P]
    gemm_kernel<0><<<dim3(2, 1, B_), 256, 0, stream>>>(
        v_w, vf, vmat, nullptr, nullptr, nullptr, VD_, P_, C_);
    // 5) q projection: [512x640] x [640x1024]
    gemm_kernel<0><<<dim3(8, 4, B_), 256, 0, stream>>>(
        q_w, xn, qproj, nullptr, nullptr, nullptr, 512, L_, C_);
    // 6) attention
    attn_kernel<<<dim3(16, NH_, B_), 128, 0, stream>>>(qproj, kmat, vmat, omat);
    // 7) out projection + layer scale + residual -> fp32 d_out
    gemm_kernel<2><<<dim3(8, 5, B_), 256, 0, stream>>>(
        out_w, omat, nullptr, out, ls, x, C_, L_, 512);
}
